// MultiHeadAttention_75024488727231
// MI455X (gfx1250) — compile-verified, matrix operand order checked by reading, not run on hardware
//
#include <hip/hip_runtime.h>
#include <cstdint>
#include <cstddef>

// ---------------------------------------------------------------------------
// Types for CDNA5 WMMA (gfx1250): v_wmma_f32_16x16x32_bf16
// ---------------------------------------------------------------------------
typedef __attribute__((ext_vector_type(16))) __bf16 bf16x16;
typedef __attribute__((ext_vector_type(8)))  __bf16 bf16x8;
typedef __attribute__((ext_vector_type(8)))  float  f32x8;

#if __has_builtin(__builtin_amdgcn_exp2f)
#define EXP2F(x) __builtin_amdgcn_exp2f(x)
#else
#define EXP2F(x) exp2f(x)
#endif
#if __has_builtin(__builtin_amdgcn_rcpf)
#define RCPF(x) __builtin_amdgcn_rcpf(x)
#else
#define RCPF(x) (1.0f / (x))
#endif

// Scheduler pinning (LLVM AMDGPU): masks — 0x008 MFMA/WMMA, 0x020 VMEM read.
#if __has_builtin(__builtin_amdgcn_sched_group_barrier)
#define SCHED_VMEM_READ(n) __builtin_amdgcn_sched_group_barrier(0x020, (n), 0)
#define SCHED_WMMA(n)      __builtin_amdgcn_sched_group_barrier(0x008, (n), 0)
#else
#define SCHED_VMEM_READ(n)
#define SCHED_WMMA(n)
#endif
#if __has_builtin(__builtin_amdgcn_sched_barrier)
// allow ALU(0x1) / DS(0x80) / transcendental(0x400) to cross; pin VMEM & WMMA
#define SCHED_FENCE_VMEM() __builtin_amdgcn_sched_barrier(0x1 | 0x80 | 0x400)
#else
#define SCHED_FENCE_VMEM()
#endif

// softmax scale folded into q: (1/sqrt(64)) * log2(e)
#define Q_SCALE 0.18033688011112042f
#define NEG_BIG (-1.0e30f)

static __device__ __forceinline__ bf16x8 ldg8(const __bf16* p) {
  return *reinterpret_cast<const bf16x8*>(p);
}
static __device__ __forceinline__ bf16x16 join8(bf16x8 a, bf16x8 b) {
  return __builtin_shufflevector(a, b, 0,1,2,3,4,5,6,7,8,9,10,11,12,13,14,15);
}
static __device__ __forceinline__ f32x8 wmma_bf16(bf16x16 a, bf16x16 b, f32x8 c) {
  // D = A(16xK=32) * B(K=32x16) + C, f32 accum, no neg/abs, no reuse hints
  return __builtin_amdgcn_wmma_f32_16x16x32_bf16(false, a, false, b,
                                                 (short)0, c, false, false);
}

// ---------------------------------------------------------------------------
// Problem constants
// ---------------------------------------------------------------------------
#define BB     4
#define TT     2048
#define CC     1024
#define HH     16
#define DD     64
#define MROWS  (BB * TT)          // 8192
#define N_QKV  (3 * CC)           // 3072

// ---------------------------------------------------------------------------
// Kernel 0a: fp32 -> bf16 convert
// ---------------------------------------------------------------------------
__global__ void cvt_f32_bf16_kernel(const float* __restrict__ in,
                                    __bf16* __restrict__ out, int n) {
  int i = blockIdx.x * blockDim.x + threadIdx.x;
  if (i < n) out[i] = (__bf16)in[i];
}

// Kernel 0b: transpose [R][C] fp32 -> [C][R] bf16
__global__ void transpose_f32_bf16_kernel(const float* __restrict__ in,
                                          __bf16* __restrict__ out,
                                          int R, int C) {
  int i = blockIdx.x * blockDim.x + threadIdx.x;
  if (i < R * C) {
    int r = i / C, c = i - r * C;
    out[(size_t)c * R + r] = (__bf16)in[i];
  }
}

// ---------------------------------------------------------------------------
// Shared GEMM inner loop: 16(M) x 64(N) per wave, K steps of 32.
// Distance-2 software pipeline with compile-time parity buffers and a FULLY
// UNROLLED K loop: iteration i consumes fragments loaded at i-2 and issues
// loads for i+2.  Between any load and its consuming WMMA sit the 4 WMMAs and
// 10 loads of the intervening iteration, so waits are partial by construction.
// ---------------------------------------------------------------------------
template<int K>
static __device__ __forceinline__ void gemm_16x64(const __bf16* __restrict__ arow,
                                                  const __bf16* const brow[4],
                                                  f32x8 acc[4]) {
  constexpr int NIT = K / 32;
  static_assert(NIT >= 2, "K must cover at least 2 pipeline stages");

  bf16x16 abuf[2];
  bf16x16 bbuf[2][4];

  // prologue: stages 0 and 1 in flight
#pragma unroll
  for (int i = 0; i < 2; ++i) {
    const int kk = i * 32;
    abuf[i] = join8(ldg8(arow + kk), ldg8(arow + kk + 16));
#pragma unroll
    for (int u = 0; u < 4; ++u)
      bbuf[i][u] = join8(ldg8(brow[u] + kk), ldg8(brow[u] + kk + 16));
  }

#pragma unroll
  for (int i = 0; i < NIT; ++i) {
    const int p = i & 1;
#pragma unroll
    for (int u = 0; u < 4; ++u)
      acc[u] = wmma_bf16(abuf[p], bbuf[p][u], acc[u]);

    if (i + 2 < NIT) {
      const int kk = (i + 2) * 32;
      abuf[p] = join8(ldg8(arow + kk), ldg8(arow + kk + 16));
#pragma unroll
      for (int u = 0; u < 4; ++u)
        bbuf[p][u] = join8(ldg8(brow[u] + kk), ldg8(brow[u] + kk + 16));

      // interleave this iteration's 4 WMMAs with the 10 prefetch loads
      SCHED_WMMA(1);
      SCHED_VMEM_READ(3);
      SCHED_WMMA(1);
      SCHED_VMEM_READ(3);
      SCHED_WMMA(1);
      SCHED_VMEM_READ(2);
      SCHED_WMMA(1);
      SCHED_VMEM_READ(2);
    }
  }
}

// ---------------------------------------------------------------------------
// Kernel 1: QKV GEMM.  xb:[M][K] bf16, wt:[N][K] bf16 (pre-transposed).
// Scatter: q -> [BH][T][D] (scaled), k -> [BH][T][D], v -> [BH][D][T].
// min-waves-per-EU = 1 relaxes the RA pressure target so the pipeline
// buffers are not collapsed.
// ---------------------------------------------------------------------------
__global__ __launch_bounds__(256, 1)
void qkv_gemm_kernel(const __bf16* __restrict__ xb,
                     const __bf16* __restrict__ wt,
                     const float*  __restrict__ bias,
                     __bf16* __restrict__ qo,
                     __bf16* __restrict__ ko,
                     __bf16* __restrict__ vto) {
  const int K = CC;
  const int lane = threadIdx.x & 31;
  const int wv   = threadIdx.x >> 5;
  const int w    = blockIdx.x * (blockDim.x >> 5) + wv;
  const int nBlk = N_QKV / 64;                 // 48
  const int mT   = w / nBlk;
  const int nB   = w - mT * nBlk;
  const int m0   = mT * 16;
  const int n0   = nB * 64;
  const int l15  = lane & 15;
  const int hi8  = (lane >> 4) << 3;           // 0 or 8

  const __bf16* arow = xb + (size_t)(m0 + l15) * K + hi8;
  const __bf16* brow[4];
#pragma unroll
  for (int u = 0; u < 4; ++u)
    brow[u] = wt + (size_t)(n0 + u * 16 + l15) * K + hi8;

  f32x8 acc[4] = {};
  gemm_16x64<CC>(arow, brow, acc);

#pragma unroll
  for (int u = 0; u < 4; ++u) {
    const int n   = n0 + u * 16 + l15;
    const float bv = bias[n];
    const int sec = n >> 10;                   // 0=q 1=k 2=v (uniform per tile)
    const int nn  = n & (CC - 1);
    const int h   = nn >> 6;
    const int d   = nn & (DD - 1);
#pragma unroll
    for (int j = 0; j < 8; ++j) {
      const int gm = m0 + j + hi8;             // global row in [0, M)
      const int b  = gm >> 11;                 // / T
      const int t  = gm & (TT - 1);
      const size_t bh = (size_t)(b * HH + h);
      const float v = acc[u][j] + bv;
      if (sec == 0)      qo[(bh * TT + t) * DD + d] = (__bf16)(v * Q_SCALE);
      else if (sec == 1) ko[(bh * TT + t) * DD + d] = (__bf16)v;
      else               vto[(bh * DD + d) * TT + t] = (__bf16)v;
    }
  }
}

// ---------------------------------------------------------------------------
// Kernel 2: causal flash attention. One wave per 16-row q tile.
// q:[BH][T][D] (pre-scaled to exp2 domain), k:[BH][T][D], vt:[BH][D][T].
// All K0 and V fragment loads for a 32-key chunk are issued (and sched-fenced)
// before the first WMMA, so V latency hides behind S-WMMAs + softmax + LDS.
// ---------------------------------------------------------------------------
__global__ __launch_bounds__(128)
void attn_kernel(const __bf16* __restrict__ q,
                 const __bf16* __restrict__ k,
                 const __bf16* __restrict__ vt,
                 __bf16* __restrict__ y) {
  const int lane = threadIdx.x & 31;
  const int wv   = threadIdx.x >> 5;
  const int wid  = blockIdx.x * (blockDim.x >> 5) + wv;   // 0 .. BH*128-1
  const int qT   = wid & 127;                             // T/16 tiles
  const int bh   = wid >> 7;
  const int qRow = qT * 16;
  const int l15  = lane & 15;
  const int hi8  = (lane >> 4) << 3;

  __shared__ __align__(16) __bf16 pbuf[4][16][40];        // 80B row stride
  __bf16 (*prow)[40] = pbuf[wv];

  // Q fragments (held for whole key loop)
  const __bf16* qb = q + ((size_t)bh * TT + qRow + l15) * DD + hi8;
  const bf16x16 aq0 = join8(ldg8(qb),      ldg8(qb + 16));   // d 0..31
  const bf16x16 aq1 = join8(ldg8(qb + 32), ldg8(qb + 48));   // d 32..63

  float mrun[8], lrun[8];
  f32x8 o[4] = {};
#pragma unroll
  for (int j = 0; j < 8; ++j) { mrun[j] = NEG_BIG; lrun[j] = 0.0f; }

  const int kEnd = qRow + 16;                    // exclusive key bound
  for (int kc = 0; kc < kEnd; kc += 32) {
    const bool t1 = (kc + 16) < kEnd;            // wave-uniform

    // ---- issue ALL chunk loads up front ----
    const __bf16* kb0 = k + ((size_t)bh * TT + kc + l15) * DD + hi8;
    const bf16x16 bk00 = join8(ldg8(kb0),      ldg8(kb0 + 16));   // keys 0-15, d 0-31
    const bf16x16 bk01 = join8(ldg8(kb0 + 32), ldg8(kb0 + 48));   // keys 0-15, d 32-63
    bf16x16 bv[4];
#pragma unroll
    for (int t = 0; t < 4; ++t) {
      const __bf16* vb = vt + ((size_t)bh * DD + t * 16 + l15) * TT + kc + hi8;
      bv[t] = join8(ldg8(vb), ldg8(vb + 16));
    }
    SCHED_FENCE_VMEM();   // keep these 12 loads issued before any WMMA below

    // ---- S = Q @ K^T (exp2 domain, scale folded into q) ----
    f32x8 s0 = {};
    s0 = wmma_bf16(aq0, bk00, s0);
    s0 = wmma_bf16(aq1, bk01, s0);
    f32x8 s1;
    if (t1) {
      const __bf16* kb1 = kb0 + 16 * DD;
      const bf16x16 bk10 = join8(ldg8(kb1),      ldg8(kb1 + 16));
      const bf16x16 bk11 = join8(ldg8(kb1 + 32), ldg8(kb1 + 48));
      s1 = (f32x8){};
      s1 = wmma_bf16(aq0, bk10, s1);
      s1 = wmma_bf16(aq1, bk11, s1);
    } else {
#pragma unroll
      for (int j = 0; j < 8; ++j) s1[j] = NEG_BIG;
    }

    // ---- causal mask (only straddling tiles need it) ----
    if (kc + 15 > qRow) {
#pragma unroll
      for (int j = 0; j < 8; ++j)
        s0[j] = (kc + l15 <= qRow + j + hi8) ? s0[j] : NEG_BIG;
    }
    if (t1 && (kc + 31 > qRow)) {
#pragma unroll
      for (int j = 0; j < 8; ++j)
        s1[j] = (kc + 16 + l15 <= qRow + j + hi8) ? s1[j] : NEG_BIG;
    }

    // ---- online softmax update (exp2 domain) ----
    float alpha[8];
#pragma unroll
    for (int j = 0; j < 8; ++j) {
      float tm = fmaxf(s0[j], s1[j]);
#pragma unroll
      for (int dx = 1; dx < 16; dx <<= 1)
        tm = fmaxf(tm, __shfl_xor(tm, dx, 16));
      const float nm = fmaxf(mrun[j], tm);
      const float al = EXP2F(mrun[j] - nm);
      mrun[j] = nm;
      const float p0 = EXP2F(s0[j] - nm);
      const float p1 = EXP2F(s1[j] - nm);
      lrun[j] = lrun[j] * al + p0 + p1;          // per-lane partial sum
      alpha[j] = al;
      prow[j + hi8][l15]      = (__bf16)p0;      // C-layout -> LDS row-major
      prow[j + hi8][16 + l15] = (__bf16)p1;
    }
#pragma unroll
    for (int t = 0; t < 4; ++t)
#pragma unroll
      for (int j = 0; j < 8; ++j) o[t][j] *= alpha[j];

    // ---- reload P in A-layout, then O += P @ V ----
    const __bf16* pl = &prow[l15][hi8];
    const bf16x16 ap = join8(*reinterpret_cast<const bf16x8*>(pl),
                             *reinterpret_cast<const bf16x8*>(pl + 16));
#pragma unroll
    for (int t = 0; t < 4; ++t)
      o[t] = wmma_bf16(ap, bv[t], o[t]);
  }

  // ---- finalize: reduce l across lanes, normalize, store y ----
  float inv[8];
#pragma unroll
  for (int j = 0; j < 8; ++j) {
    float s = lrun[j];
#pragma unroll
    for (int dx = 1; dx < 16; dx <<= 1) s += __shfl_xor(s, dx, 16);
    inv[j] = RCPF(s);
  }
  const int b = bh >> 4, h = bh & (HH - 1);
#pragma unroll
  for (int t = 0; t < 4; ++t)
#pragma unroll
    for (int j = 0; j < 8; ++j) {
      const size_t row = (size_t)(b * TT + qRow + j + hi8);
      y[row * CC + h * DD + t * 16 + l15] = (__bf16)(o[t][j] * inv[j]);
    }
}

// ---------------------------------------------------------------------------
// Kernel 3: output projection. yb:[M][K] bf16, wt:[N][K] bf16, out fp32 [M][N].
// ---------------------------------------------------------------------------
__global__ __launch_bounds__(256, 1)
void proj_gemm_kernel(const __bf16* __restrict__ yb,
                      const __bf16* __restrict__ wt,
                      const float*  __restrict__ bias,
                      float* __restrict__ out) {
  const int K = CC, N = CC;
  const int lane = threadIdx.x & 31;
  const int wv   = threadIdx.x >> 5;
  const int w    = blockIdx.x * (blockDim.x >> 5) + wv;
  const int nBlk = N / 64;                      // 16
  const int mT   = w / nBlk;
  const int nB   = w - mT * nBlk;
  const int m0   = mT * 16;
  const int n0   = nB * 64;
  const int l15  = lane & 15;
  const int hi8  = (lane >> 4) << 3;

  const __bf16* arow = yb + (size_t)(m0 + l15) * K + hi8;
  const __bf16* brow[4];
#pragma unroll
  for (int u = 0; u < 4; ++u)
    brow[u] = wt + (size_t)(n0 + u * 16 + l15) * K + hi8;

  f32x8 acc[4] = {};
  gemm_16x64<CC>(arow, brow, acc);

#pragma unroll
  for (int u = 0; u < 4; ++u) {
    const int n = n0 + u * 16 + l15;
    const float bv = bias[n];
#pragma unroll
    for (int j = 0; j < 8; ++j) {
      const size_t gm = (size_t)(m0 + j + hi8);
      out[gm * N + n] = acc[u][j] + bv;
    }
  }
}

// ---------------------------------------------------------------------------
// Launch
// ---------------------------------------------------------------------------
extern "C" void kernel_launch(void* const* d_in, const int* in_sizes, int n_in,
                              void* d_out, int out_size, void* d_ws, size_t ws_size,
                              hipStream_t stream) {
  const float* x      = (const float*)d_in[0];   // [B,T,C]
  const float* w_attn = (const float*)d_in[1];   // [C, 3C]
  const float* b_attn = (const float*)d_in[2];   // [3C]
  const float* w_proj = (const float*)d_in[3];   // [C, C]
  const float* b_proj = (const float*)d_in[4];   // [C]
  float* out = (float*)d_out;

  char* ws = (char*)d_ws;
  const size_t MB = 1u << 20;
  __bf16* xb   = (__bf16*)(ws);                  // 16 MB  [M][C]
  __bf16* wta  = (__bf16*)(ws + 16 * MB);        //  6 MB  [3C][C]
  __bf16* wtp  = (__bf16*)(ws + 22 * MB);        //  2 MB  [C][C]
  __bf16* qws  = (__bf16*)(ws + 24 * MB);        // 16 MB  [BH][T][D]
  __bf16* kws  = (__bf16*)(ws + 40 * MB);        // 16 MB  [BH][T][D]
  __bf16* vtws = (__bf16*)(ws + 56 * MB);        // 16 MB  [BH][D][T]
  __bf16* yws  = xb;                             // reuse: x dead after QKV GEMM

  // 0) convert / transpose to bf16
  {
    int n = MROWS * CC;                          // 8388608
    cvt_f32_bf16_kernel<<<(n + 255) / 256, 256, 0, stream>>>(x, xb, n);
  }
  {
    int n = CC * N_QKV;                          // 3145728
    transpose_f32_bf16_kernel<<<(n + 255) / 256, 256, 0, stream>>>(w_attn, wta, CC, N_QKV);
  }
  {
    int n = CC * CC;                             // 1048576
    transpose_f32_bf16_kernel<<<(n + 255) / 256, 256, 0, stream>>>(w_proj, wtp, CC, CC);
  }

  // 1) QKV GEMM + scatter into attention layouts
  {
    int waves  = (MROWS / 16) * (N_QKV / 64);    // 24576
    int blocks = waves / 8;                      // 256 threads = 8 waves
    qkv_gemm_kernel<<<blocks, 256, 0, stream>>>(xb, wta, b_attn, qws, kws, vtws);
  }

  // 2) causal flash attention
  {
    int waves  = (BB * HH) * (TT / 16);          // 8192
    int blocks = waves / 4;                      // 128 threads = 4 waves
    attn_kernel<<<blocks, 128, 0, stream>>>(qws, kws, vtws, yws);
  }

  // 3) output projection -> fp32 d_out
  {
    int waves  = (MROWS / 16) * (CC / 64);       // 8192
    int blocks = waves / 8;
    proj_gemm_kernel<<<blocks, 256, 0, stream>>>(yws, wtp, b_proj, out);
  }

  (void)in_sizes; (void)n_in; (void)out_size; (void)ws_size;
}
